// PWSA_Layer_58007828300557
// MI455X (gfx1250) — matrix operand order repaired and checked by reading, not compiled, not run on hardware
//
#include <hip/hip_runtime.h>
#include <hip/hip_bf16.h>

// PWSA layer, MI455X (gfx1250, wave32, WMMA bf16 path).
// energy = (K^T K)/sqrt(D) with tied q/k weights; two-pass softmax
// (stats then recompute+accumulate), all matmuls on v_wmma_f32_16x16x32_bf16.

#define C_DIM 512
#define N_DIM 2048
#define D_DIM 128
#define XPAD  520   // LDS x-tile row stride (bf16 elems), mult of 8
#define PTPAD 40    // LDS P-tile row stride (bf16 elems), mult of 8

typedef __attribute__((ext_vector_type(16))) __bf16 v16bf;
typedef __attribute__((ext_vector_type(8)))  __bf16 v8bf;
typedef __attribute__((ext_vector_type(8)))  float  v8f;

static __device__ __forceinline__ v16bf ld16(const __bf16* p0, const __bf16* p1) {
  v8bf a = *(const v8bf*)p0;
  v8bf b = *(const v8bf*)p1;
  return __builtin_shufflevector(a, b, 0,1,2,3,4,5,6,7,8,9,10,11,12,13,14,15);
}

static __device__ __forceinline__ v8f wmma_bf16(v16bf a, v16bf b, v8f c) {
  return __builtin_amdgcn_wmma_f32_16x16x32_bf16(
      /*neg_a=*/false, a, /*neg_b=*/false, b,
      /*c_mod=*/(short)0, c, /*reuse_a=*/false, /*reuse_b=*/false);
}

// ---------------- kernel 0: weights f32 -> bf16 -------------------------
__global__ __launch_bounds__(256) void cvt_w_kernel(
    const float* __restrict__ wqk, const float* __restrict__ wv,
    __bf16* __restrict__ wqk_bf, __bf16* __restrict__ wv_bf) {
  int i = blockIdx.x * 256 + threadIdx.x;
  if (i < D_DIM * C_DIM) {
    wqk_bf[i] = (__bf16)wqk[i];
    wv_bf[i]  = (__bf16)wv[i];
  }
}

// ---------------- kernel 1: projections (WMMA) --------------------------
// grid (N/32, B), 256 threads. Waves 0-3: K = wqk*x stored [B][N][D].
// Waves 4-7: V = wv*x + bv stored transposed [B][D][N].
__global__ __launch_bounds__(256) void proj_kernel(
    const float* __restrict__ x, const __bf16* __restrict__ wqk,
    const __bf16* __restrict__ wv, const float* __restrict__ bv,
    __bf16* __restrict__ Kbf, __bf16* __restrict__ Vt) {
  __shared__ __align__(16) __bf16 xt[32 * XPAD];  // x^T tile [n][c], bf16
  const int b    = blockIdx.y;
  const int n0   = blockIdx.x * 32;
  const int t    = threadIdx.x;
  const int lane = t & 31, wave = t >> 5;
  const int half = lane >> 4, lm = lane & 15;

  // Load + transpose x[:, n0:n0+32] (coalesced along n) into LDS as [n][c].
  {
    const float* xb = x + (size_t)b * C_DIM * N_DIM + n0;
    const int n = t & 31;
    const int cb = t >> 5;
#pragma unroll 4
    for (int it = 0; it < 64; ++it) {
      int c = cb + it * 8;
      xt[n * XPAD + c] = (__bf16)xb[(size_t)c * N_DIM + n];
    }
  }
  __syncthreads();

  if (wave < 4) {
    // K projection: M=d, N=n, K-dim=c.  out[d,n] = sum_c wqk[d,c] x^T[n,c]
#pragma unroll
    for (int dq = 0; dq < 2; ++dq) {
      const int dt = wave * 2 + dq;
      const __bf16* wrow = wqk + (size_t)(dt * 16 + lm) * C_DIM;
#pragma unroll
      for (int nt = 0; nt < 2; ++nt) {
        const __bf16* xrow = xt + (nt * 16 + lm) * XPAD;
        v8f acc = {};
#pragma unroll
        for (int cb = 0; cb < 16; ++cb) {
          const int c0 = cb * 32;
          v16bf a  = ld16(wrow + c0 + half * 8, wrow + c0 + 16 + half * 8);
          v16bf bb = ld16(xrow + c0 + half * 8, xrow + c0 + 16 + half * 8);
          acc = wmma_bf16(a, bb, acc);
        }
        // D-tile: lane column n = nt*16+lm, rows d = dt*16 + 8*half + r
        v8bf o;
#pragma unroll
        for (int r = 0; r < 8; ++r) o[r] = (__bf16)acc[r];
        __bf16* dst = Kbf + ((size_t)(b * N_DIM + n0 + nt * 16 + lm) * D_DIM
                             + dt * 16 + half * 8);
        *(v8bf*)dst = o;
      }
    }
  } else {
    // V projection: M=n, N=d, K-dim=c.  out[n,d] = sum_c x^T[n,c] wv[d,c] + bv[d]
#pragma unroll
    for (int dq = 0; dq < 2; ++dq) {
      const int dt = (wave - 4) * 2 + dq;
      const float bias = bv[dt * 16 + lm];
      const __bf16* wrow = wv + (size_t)(dt * 16 + lm) * C_DIM;
#pragma unroll
      for (int nt = 0; nt < 2; ++nt) {
        const __bf16* xrow = xt + (nt * 16 + lm) * XPAD;
        v8f acc;
#pragma unroll
        for (int r = 0; r < 8; ++r) acc[r] = bias;
#pragma unroll
        for (int cb = 0; cb < 16; ++cb) {
          const int c0 = cb * 32;
          v16bf a  = ld16(xrow + c0 + half * 8, xrow + c0 + 16 + half * 8);
          v16bf bb = ld16(wrow + c0 + half * 8, wrow + c0 + 16 + half * 8);
          acc = wmma_bf16(a, bb, acc);
        }
        // D-tile: lane column d = dt*16+lm, rows n = n0 + nt*16 + 8*half + r
        v8bf o;
#pragma unroll
        for (int r = 0; r < 8; ++r) o[r] = (__bf16)acc[r];
        __bf16* dst = Vt + ((size_t)(b * D_DIM + dt * 16 + lm) * N_DIM
                            + n0 + nt * 16 + half * 8);
        *(v8bf*)dst = o;
      }
    }
  }
}

// ---------------- kernel 2: softmax row stats (WMMA + online max/sum) ---
// grid (N/32, B), 256 threads. WG owns 32 rows n; wave w streams m strips.
__global__ __launch_bounds__(256) void stats_kernel(
    const __bf16* __restrict__ Kbf, float* __restrict__ Mrow,
    float* __restrict__ Lrow) {
  __shared__ float smax[8][32];
  __shared__ float ssum[8][32];
  const int b    = blockIdx.y;
  const int n0   = blockIdx.x * 32;
  const int t    = threadIdx.x;
  const int lane = t & 31, wave = t >> 5;
  const int half = lane >> 4, lm = lane & 15;
  const float invScale = 0.08838834764831845f;  // 1/sqrt(128)

  const __bf16* Kb = Kbf + (size_t)b * N_DIM * D_DIM;

  // Preload A: K rows n (2 n-subtiles x 4 d-chunks)
  v16bf a[2][4];
#pragma unroll
  for (int ns = 0; ns < 2; ++ns) {
    const __bf16* row = Kb + (size_t)(n0 + ns * 16 + lm) * D_DIM;
#pragma unroll
    for (int dc = 0; dc < 4; ++dc)
      a[ns][dc] = ld16(row + dc * 32 + half * 8, row + dc * 32 + 16 + half * 8);
  }

  float rmax[2][8], rsum[2][8];
#pragma unroll
  for (int ns = 0; ns < 2; ++ns)
#pragma unroll
    for (int r = 0; r < 8; ++r) { rmax[ns][r] = -INFINITY; rsum[ns][r] = 0.f; }

  for (int it = 0; it < 8; ++it) {
    const int m0 = it * 256 + wave * 32;
    v16bf bb[2][4];
#pragma unroll
    for (int ms = 0; ms < 2; ++ms) {
      const __bf16* row = Kb + (size_t)(m0 + ms * 16 + lm) * D_DIM;
#pragma unroll
      for (int dc = 0; dc < 4; ++dc)
        bb[ms][dc] = ld16(row + dc * 32 + half * 8, row + dc * 32 + 16 + half * 8);
    }
#pragma unroll
    for (int ns = 0; ns < 2; ++ns) {
      v8f s0 = {}, s1 = {};
#pragma unroll
      for (int dc = 0; dc < 4; ++dc) {
        s0 = wmma_bf16(a[ns][dc], bb[0][dc], s0);
        s1 = wmma_bf16(a[ns][dc], bb[1][dc], s1);
      }
      // Online stats; row (ns, half, r) lives across 16 lanes of this half.
#pragma unroll
      for (int r = 0; r < 8; ++r) {
        float v0 = s0[r] * invScale;
        float v1 = s1[r] * invScale;
        float tm = fmaxf(v0, v1);
#pragma unroll
        for (int m = 1; m < 16; m <<= 1) tm = fmaxf(tm, __shfl_xor(tm, m));
        float nm = fmaxf(rmax[ns][r], tm);
        float es = __expf(v0 - nm) + __expf(v1 - nm);
#pragma unroll
        for (int m = 1; m < 16; m <<= 1) es += __shfl_xor(es, m);
        rsum[ns][r] = rsum[ns][r] * __expf(rmax[ns][r] - nm) + es;
        rmax[ns][r] = nm;
      }
    }
  }
  if (lm == 0) {
#pragma unroll
    for (int ns = 0; ns < 2; ++ns)
#pragma unroll
      for (int r = 0; r < 8; ++r) {
        int row = ns * 16 + half * 8 + r;
        smax[wave][row] = rmax[ns][r];
        ssum[wave][row] = rsum[ns][r];
      }
  }
  __syncthreads();
  if (t < 32) {
    float gm = -INFINITY;
#pragma unroll
    for (int w = 0; w < 8; ++w) gm = fmaxf(gm, smax[w][t]);
    float gs = 0.f;
#pragma unroll
    for (int w = 0; w < 8; ++w) gs += ssum[w][t] * __expf(smax[w][t] - gm);
    Mrow[(size_t)b * N_DIM + n0 + t] = gm;
    Lrow[(size_t)b * N_DIM + n0 + t] = gs;
  }
}

// ---------------- kernel 3: output O[d,m] = sum_n V[d,n] P[n,m] ---------
// grid (N/128, B), 256 threads; wave owns 16 output columns m.
__global__ __launch_bounds__(256) void attn_out_kernel(
    const __bf16* __restrict__ Kbf, const __bf16* __restrict__ Vt,
    const float* __restrict__ Mrow, const float* __restrict__ Lrow,
    float* __restrict__ out) {
  __shared__ __align__(16) __bf16 pt[8][16 * PTPAD];  // per-wave P^T tile [m][n]
  const int b    = blockIdx.y;
  const int t    = threadIdx.x;
  const int lane = t & 31, wave = t >> 5;
  const int half = lane >> 4, lm = lane & 15;
  const int m0   = blockIdx.x * 128 + wave * 16;
  const float invScale = 0.08838834764831845f;

  const __bf16* Kb = Kbf + (size_t)b * N_DIM * D_DIM;
  const __bf16* Vb = Vt + (size_t)b * D_DIM * N_DIM;
  __bf16* ptw = &pt[wave][0];

  // Preload B_m: K columns m (fixed for this wave), 4 d-chunks.
  v16bf bm[4];
  {
    const __bf16* row = Kb + (size_t)(m0 + lm) * D_DIM;
#pragma unroll
    for (int dc = 0; dc < 4; ++dc)
      bm[dc] = ld16(row + dc * 32 + half * 8, row + dc * 32 + 16 + half * 8);
  }

  v8f acco[8];
  {
    v8f z = {};
#pragma unroll
    for (int dt = 0; dt < 8; ++dt) acco[dt] = z;
  }

  for (int nch = 0; nch < 64; ++nch) {
    const int nb = nch * 32;
    // Row stats: lane l caches row nb+l; broadcast via shuffle below.
    const float mrow_l = Mrow[(size_t)b * N_DIM + nb + lane];
    const float linv_l = 1.0f / Lrow[(size_t)b * N_DIM + nb + lane];

#pragma unroll
    for (int ns = 0; ns < 2; ++ns) {
      const __bf16* row = Kb + (size_t)(nb + ns * 16 + lm) * D_DIM;
      v8f s = {};
#pragma unroll
      for (int dc = 0; dc < 4; ++dc) {
        v16bf a = ld16(row + dc * 32 + half * 8, row + dc * 32 + 16 + half * 8);
        s = wmma_bf16(a, bm[dc], s);
      }
      // P = exp(S/scale - M[n]) / L[n]; store transposed tile [m][n] in LDS.
      v8bf o;
#pragma unroll
      for (int r = 0; r < 8; ++r) {
        int ridx = ns * 16 + half * 8 + r;
        float mv = __shfl(mrow_l, ridx);
        float lv = __shfl(linv_l, ridx);
        o[r] = (__bf16)(__expf(s[r] * invScale - mv) * lv);
      }
      *(v8bf*)(ptw + lm * PTPAD + ns * 16 + half * 8) = o;
    }
    // Per-wave LDS ops are in-order; load P^T as B operand (k = n).
    v16bf pb = ld16(ptw + lm * PTPAD + half * 8, ptw + lm * PTPAD + 16 + half * 8);
#pragma unroll
    for (int dt = 0; dt < 8; ++dt) {
      const __bf16* vrow = Vb + (size_t)(dt * 16 + lm) * N_DIM + nb;
      v16bf a = ld16(vrow + half * 8, vrow + 16 + half * 8);
      acco[dt] = wmma_bf16(a, pb, acco[dt]);
    }
  }

  // Store: D-tile lane column m = m0+lm, rows d = dt*16 + 8*half + r.
  float* ob = out + (size_t)b * D_DIM * N_DIM;
#pragma unroll
  for (int dt = 0; dt < 8; ++dt)
#pragma unroll
    for (int r = 0; r < 8; ++r) {
      int d = dt * 16 + half * 8 + r;
      ob[(size_t)d * N_DIM + m0 + lm] = acco[dt][r];
    }
}

// ---------------- launcher ----------------------------------------------
extern "C" void kernel_launch(void* const* d_in, const int* in_sizes, int n_in,
                              void* d_out, int out_size, void* d_ws, size_t ws_size,
                              hipStream_t stream) {
  const float* x   = (const float*)d_in[0];
  const float* wqk = (const float*)d_in[1];
  const float* wv  = (const float*)d_in[2];
  const float* bv  = (const float*)d_in[3];
  float* out = (float*)d_out;

  char* ws = (char*)d_ws;
  __bf16* Kbf    = (__bf16*)(ws + 0);         // [B][N][D] bf16, 4 MB
  __bf16* Vt     = (__bf16*)(ws + 4194304);   // [B][D][N] bf16, 4 MB
  float*  Mrow   = (float*)(ws + 8388608);    // [B][N] row max, 64 KB
  float*  Lrow   = (float*)(ws + 8454144);    // [B][N] row sum, 64 KB
  __bf16* wqk_bf = (__bf16*)(ws + 8519680);   // 128 KB
  __bf16* wv_bf  = (__bf16*)(ws + 8650752);   // 128 KB

  cvt_w_kernel<<<256, 256, 0, stream>>>(wqk, wv, wqk_bf, wv_bf);
  proj_kernel<<<dim3(64, 8), 256, 0, stream>>>(x, wqk_bf, wv_bf, bv, Kbf, Vt);
  stats_kernel<<<dim3(64, 8), 256, 0, stream>>>(Kbf, Mrow, Lrow);
  attn_out_kernel<<<dim3(16, 8), 256, 0, stream>>>(Kbf, Vt, Mrow, Lrow, out);
}
